// H2GCNConv_68143951118621
// MI455X (gfx1250) — matrix operand order repaired and checked by reading, not compiled, not run on hardware
//
#include <hip/hip_runtime.h>

typedef __attribute__((ext_vector_type(16))) _Float16 v16h;
typedef __attribute__((ext_vector_type(8)))  float    v8f;

#define DFEAT 64

// ---------------------------------------------------------------------------
// Zero workspace (h1, h2) with float4 stores.
// ---------------------------------------------------------------------------
__global__ __launch_bounds__(256) void zero_f4(float4* __restrict__ p, int n4) {
    int i = blockIdx.x * 256 + threadIdx.x;
    if (i < n4) p[i] = make_float4(0.f, 0.f, 0.f, 0.f);
}

// ---------------------------------------------------------------------------
// SpMM hop: dst[row] += src[col] * w  (gather-scale-scatter-add).
// 16 threads per edge, 4 features each: float4 gather + 4 HW f32 atomics.
// Feature matrices (25.6 MB) are L2-resident on MI455X (192 MB L2), so the
// random gathers and atomics mostly stay on-chip.
// ---------------------------------------------------------------------------
__global__ __launch_bounds__(256) void spmm_atomic(
    const int*   __restrict__ rows,
    const int*   __restrict__ cols,
    const float* __restrict__ ew,
    const float* __restrict__ src,
    float*       __restrict__ dst,
    int E)
{
    int tid = blockIdx.x * 256 + threadIdx.x;
    int e = tid >> 4;
    if (e >= E) return;
    int f = (tid & 15) << 2;                       // feature offset 0..60
    int r = rows[e];
    int c = cols[e];
    float w = ew[e];
    const float4 v = *reinterpret_cast<const float4*>(src + (size_t)c * DFEAT + f);
    float* d = dst + (size_t)r * DFEAT + f;
    unsafeAtomicAdd(d + 0, v.x * w);               // global_atomic_add_f32
    unsafeAtomicAdd(d + 1, v.y * w);
    unsafeAtomicAdd(d + 2, v.z * w);
    unsafeAtomicAdd(d + 3, v.w * w);
}

// ---------------------------------------------------------------------------
// Pack one A fragment (16x32 f16, wave32, contiguous k order chosen here, matched by B):
//   lanes 0-15 : row M = lane,    slots 0-15 -> K=kb+0..15
//   lanes 16-31: row M = lane-16, slots 0-15 -> K=kb+16..31
// rp points at the lane's row (float4 granularity); b = (hi?4:0) + (kstep?8:0).
// ---------------------------------------------------------------------------
__device__ __forceinline__ v16h pack_a(const float4* __restrict__ rp, int b) {
    float4 r0 = rp[b + 0];
    float4 r1 = rp[b + 1];
    float4 r2 = rp[b + 2];
    float4 r3 = rp[b + 3];
    v16h a;
    a[0]  = (_Float16)r0.x; a[1]  = (_Float16)r0.y; a[2]  = (_Float16)r0.z; a[3]  = (_Float16)r0.w;
    a[4]  = (_Float16)r1.x; a[5]  = (_Float16)r1.y; a[6]  = (_Float16)r1.z; a[7]  = (_Float16)r1.w;
    a[8]  = (_Float16)r2.x; a[9]  = (_Float16)r2.y; a[10] = (_Float16)r2.z; a[11] = (_Float16)r2.w;
    a[12] = (_Float16)r3.x; a[13] = (_Float16)r3.y; a[14] = (_Float16)r3.z; a[15] = (_Float16)r3.w;
    return a;
}

// ---------------------------------------------------------------------------
// Fused 3-hop GEMM: out[:,  0: 64] = x  @ W0^T + b0
//                   out[:, 64:128] = h1 @ W1^T + b1
//                   out[:,128:192] = h2 @ W2^T + b2
// One wave per 16-row tile; 8 waves (256 threads) per block.
// All three 64x64 weight matrices staged in LDS as f16 (24 KB).
// B fragment (32x16 f16, wave32): lane holds column N = lane%16,
//   lanes 0-15 -> K = kb+0..15, lanes 16-31 -> K = kb+16..31  (contiguous in
//   the [n][k] row-major LDS layout -> single 32B ds read).
// ---------------------------------------------------------------------------
__global__ __launch_bounds__(256) void h2gcn_gemm(
    const float* __restrict__ x,
    const float* __restrict__ h1,
    const float* __restrict__ h2,
    const float* __restrict__ W,     // [3][64][64] f32, row-major [hop][out][in]
    const float* __restrict__ bias,  // [3][64] f32
    float*       __restrict__ out,   // [N][192] f32
    int nTiles)
{
    __shared__ __align__(32) _Float16 sW[3 * 64 * 64];
    for (int i = threadIdx.x; i < 3 * 64 * 64; i += 256)
        sW[i] = (_Float16)W[i];
    __syncthreads();

    const int lane = threadIdx.x & 31;
    const int wv   = threadIdx.x >> 5;
    const int tile = blockIdx.x * 8 + wv;
    if (tile >= nTiles) return;                    // wave-uniform: EXEC all-1 for WMMA

    const int hi  = lane >> 4;                     // 0 = lanes 0-15, 1 = lanes 16-31
    const int l16 = lane & 15;
    const int row = tile * 16 + l16;               // A-matrix row held by this lane

    const float* srcs[3] = { x, h1, h2 };

    #pragma unroll
    for (int hop = 0; hop < 3; ++hop) {
        const float4* rp = reinterpret_cast<const float4*>(srcs[hop] + (size_t)row * DFEAT);
        const int ab = hi * 4;
        v16h a0 = pack_a(rp, ab);                  // K = 0..31  block
        v16h a1 = pack_a(rp, ab + 8);              // K = 32..63 block

        #pragma unroll
        for (int nt = 0; nt < 4; ++nt) {
            const int n = nt * 16 + l16;           // output column held by this lane
            const _Float16* wrow = &sW[((hop * 64 + n) * 64) + hi * 16];
            v16h bf0 = *reinterpret_cast<const v16h*>(wrow);        // K 0..31  slice
            v16h bf1 = *reinterpret_cast<const v16h*>(wrow + 32);   // K 32..63 slice

            v8f c = {};
            c = __builtin_amdgcn_wmma_f32_16x16x32_f16(false, a0, false, bf0,
                                                       (short)0, c, false, false);
            c = __builtin_amdgcn_wmma_f32_16x16x32_f16(false, a1, false, bf1,
                                                       (short)0, c, false, false);

            const float bv = bias[hop * 64 + n];
            // C/D layout: VGPR v -> M = v (lanes 0-15) or v+8 (lanes 16-31); N = lane%16
            float* op = out + (size_t)(tile * 16 + hi * 8) * 192 + hop * 64 + n;
            #pragma unroll
            for (int v = 0; v < 8; ++v)
                op[(size_t)v * 192] = c[v] + bv;
        }
    }
}

// ---------------------------------------------------------------------------
// inputs: 0:x [N*64] f32, 1:edge_index [2*E] int, 2:edge_weight [E] f32,
//         3:W [3*64*64] f32, 4:b [3*64] f32, 5:num_nodes scalar
// output: [N*192] f32. Workspace: h1 (N*64 f32) then h2 (N*64 f32) = 51.2 MB.
// ---------------------------------------------------------------------------
extern "C" void kernel_launch(void* const* d_in, const int* in_sizes, int n_in,
                              void* d_out, int out_size, void* d_ws, size_t ws_size,
                              hipStream_t stream) {
    const float* x  = (const float*)d_in[0];
    const int*   ei = (const int*)  d_in[1];
    const float* ew = (const float*)d_in[2];
    const float* W  = (const float*)d_in[3];
    const float* b  = (const float*)d_in[4];

    const int N = in_sizes[0] / DFEAT;
    const int E = in_sizes[2];

    float* h1 = (float*)d_ws;
    float* h2 = h1 + (size_t)N * DFEAT;

    // 1) zero h1 and h2
    const int n4 = (N * DFEAT * 2) / 4;
    zero_f4<<<(n4 + 255) / 256, 256, 0, stream>>>((float4*)d_ws, n4);

    // 2) hop 1: h1 = A(x);  3) hop 2: h2 = A(h1)   (stream-ordered)
    const int spmmBlocks = (int)(((size_t)E * 16 + 255) / 256);
    spmm_atomic<<<spmmBlocks, 256, 0, stream>>>(ei, ei + E, ew, x,  h1, E);
    spmm_atomic<<<spmmBlocks, 256, 0, stream>>>(ei, ei + E, ew, h1, h2, E);

    // 4) fused WMMA GEMMs + bias + concat
    const int nTiles = (N + 15) / 16;
    h2gcn_gemm<<<(nTiles + 7) / 8, 256, 0, stream>>>(x, h1, h2, W, b, (float*)d_out, nTiles);
}
